// SwinTransformerBlockPost_3204045603793
// MI455X (gfx1250) — matrix operand order repaired and checked by reading, not compiled, not run on hardware
//
#include <hip/hip_runtime.h>
#include <hip/hip_bf16.h>

typedef __attribute__((ext_vector_type(16))) _Float16 v16h;
typedef __attribute__((ext_vector_type(8)))  float    v8f;

// ---------------- geometry constants (reference: H=W=64, WS=8, SS=4) -------
#define DIMC   128
#define NTOKW  64          // tokens per window
#define XS_    130         // f16 stride: x window  [64][130]
#define QS_    34          // f16 stride: q/k      [4][64][34]
#define VS_    66          // f16 stride: v^T      [4][32][66]
#define AT_    66          // f16 stride: attn     [4][64][66]
#define AOS_   130         // f16 stride: attn out [64][130]
#define POS_   132         // f32 stride: proj out [64][132]
#define HS_    514         // f16 stride: mlp hid  [64][514]

#define QSCALE 0.17677669529663687f    // 32^-0.5

// LDS element offsets (f16 units) inside attention smem
#define OFF_XW   0
#define OFF_QB   (64 * XS_)                        // 8320
#define OFF_KB   (OFF_QB + 4 * 64 * QS_)           // 17024
#define OFF_VT   (OFF_QB + 2 * 4 * 64 * QS_)       // 25728
#define OFF_AT   (OFF_VT + 4 * 32 * VS_)
#define OFF_AO   (OFF_AT + 4 * 64 * AT_)
#define OFF_PO   (OFF_AO + 64 * AOS_)              // f32 region starts here

// ws layout (elements of _Float16 unless noted)
#define WS_QKV   0                 // 384*128 (q rows pre-scaled)
#define WS_PROJ  49152             // 128*128
#define WS_FC1   65536             // 512*128
#define WS_FC2   131072            // 128*512
#define WS_BIAS_BYTES  393216      // float [4][64][64]
#define WS_QKVB_BYTES  458752      // float [384] (q part pre-scaled)

// ---------------- WMMA fragment helpers (ISA 7.12.2 layouts) ---------------
__device__ __forceinline__ v16h frag_a(const _Float16* p, int ld, int lane) {
  // A 16x32 f16: lane m=lane&15, half=lane>>4; VGPR v -> K=(v>>2)*16 + half*8 + (v&3)*2
  const int m = lane & 15, hf = lane >> 4;
  const _Float16* row = p + m * ld;
  v16h r;
#pragma unroll
  for (int v = 0; v < 8; ++v) {
    int k = ((v >> 2) << 4) + (hf << 3) + ((v & 3) << 1);
    r[2 * v]     = row[k];
    r[2 * v + 1] = row[k + 1];
  }
  return r;
}

__device__ __forceinline__ v16h frag_b(const _Float16* p, int ld, int lane) {
  // B 32x16 f16 from row-major [n][k]: VGPR v -> K=(v>>2)*16 + half*8 + (v&3)*2 (same as A)
  const int n = lane & 15, hf = lane >> 4;
  const _Float16* row = p + n * ld;
  v16h r;
#pragma unroll
  for (int v = 0; v < 8; ++v) {
    int k = ((v >> 2) << 4) + (hf << 3) + ((v & 3) << 1);
    r[2 * v]     = row[k];
    r[2 * v + 1] = row[k + 1];
  }
  return r;
}

__device__ __forceinline__ v8f wmma16(v16h a, v16h b, v8f c) {
  return __builtin_amdgcn_wmma_f32_16x16x32_f16(false, a, false, b, (short)0, c,
                                                false, false);
}

__device__ __forceinline__ int reg3(int g) {   // shift-mask region, H=64 ws=8 ss=4
  return g < 56 ? 0 : (g < 60 ? 1 : 2);
}

// ---------------- kernel 1: weight f32 -> f16 (+ q pre-scaling) ------------
__global__ void k_convert_weights(const float* __restrict__ qkv,
                                  const float* __restrict__ proj,
                                  const float* __restrict__ fc1,
                                  const float* __restrict__ fc2,
                                  const float* __restrict__ qkvb,
                                  _Float16* __restrict__ o,
                                  float* __restrict__ qkvb_s) {
  const int i0 = blockIdx.x * blockDim.x + threadIdx.x;
  for (int i = i0; i < 196608; i += gridDim.x * blockDim.x) {
    float v, sc = 1.0f;
    if (i < 49152)      { v = qkv[i]; if (i < 16384) sc = QSCALE; }
    else if (i < 65536)   v = proj[i - 49152];
    else if (i < 131072)  v = fc1[i - 65536];
    else                  v = fc2[i - 131072];
    o[i] = (_Float16)(v * sc);
  }
  if (i0 < 384) qkvb_s[i0] = qkvb[i0] * (i0 < 128 ? QSCALE : 1.0f);
}

// ---------------- kernel 2: continuous RPE bias table ----------------------
__global__ void k_rpe_bias(const float* __restrict__ w1, const float* __restrict__ b1,
                           const float* __restrict__ w2, float* __restrict__ bias) {
  int id = blockIdx.x * 256 + threadIdx.x;       // 4096 = 64*64 (m,n) pairs
  int m = id >> 6, n = id & 63;
  int dy = (m >> 3) - (n >> 3), dx = (m & 7) - (n & 7);
  float ty = (float)dy * (8.0f / 7.0f);
  float tx = (float)dx * (8.0f / 7.0f);
  float vy = copysignf(log2f(fabsf(ty) + 1.0f) * (1.0f / 3.0f), ty);
  float vx = copysignf(log2f(fabsf(tx) + 1.0f) * (1.0f / 3.0f), tx);
  float a0 = 0.f, a1 = 0.f, a2 = 0.f, a3 = 0.f;
  for (int j = 0; j < 512; ++j) {
    float h = fmaxf(vy * w1[2 * j] + vx * w1[2 * j + 1] + b1[j], 0.0f);
    a0 += h * w2[j];
    a1 += h * w2[512 + j];
    a2 += h * w2[1024 + j];
    a3 += h * w2[1536 + j];
  }
  bias[0 * 4096 + m * 64 + n] = a0;
  bias[1 * 4096 + m * 64 + n] = a1;
  bias[2 * 4096 + m * 64 + n] = a2;
  bias[3 * 4096 + m * 64 + n] = a3;
}

// ---------------- kernel 3: fused shifted-window attention + LN + residual -
__global__ __launch_bounds__(256)
void k_attn_block(const float* __restrict__ x, const _Float16* __restrict__ w16,
                  const float* __restrict__ biasf,
                  const float* __restrict__ qkvb_s, const float* __restrict__ proj_b,
                  const float* __restrict__ n1w, const float* __restrict__ n1b,
                  float* __restrict__ out) {
  extern __shared__ char smem[];
  _Float16* smf = (_Float16*)smem;                // flat f16 view
  _Float16* xw = smf + OFF_XW;                    // [64][130]
  _Float16* qb = smf + OFF_QB;                    // [4][64][34]
  _Float16* kb = smf + OFF_KB;                    // [4][64][34]
  _Float16* vT = smf + OFF_VT;                    // [4][32][66]
  _Float16* at = smf + OFF_AT;                    // [4][64][66]
  _Float16* ao = smf + OFF_AO;                    // [64][130]
  float*    po = (float*)(smf + OFF_PO);          // [64][132]

  const int tid = threadIdx.x, lane = tid & 31, wv = tid >> 5;
  const int nlane = lane & 15, mofs = (lane >> 4) << 3;
  const int wg = blockIdx.x, b = wg >> 6, wi = wg & 63;
  const int wh = wi >> 3, ww = wi & 7;

  // ---- stage 1: gather rolled window (shift -4,-4) into LDS as f16 ----
  for (int i = tid; i < NTOKW * DIMC; i += 256) {
    int m = i >> 7, ch = i & 127;
    int gh = wh * 8 + (m >> 3), gw = ww * 8 + (m & 7);
    int oh = (gh + 4) & 63, ow = (gw + 4) & 63;
    float v = x[((long)(b * 4096 + oh * 64 + ow)) * DIMC + ch];
    xw[m * XS_ + ch] = (_Float16)v;
  }
  __syncthreads();

  // ---- stage 2: QKV GEMM 64x128 @ 128x384 (wave -> 3 ntiles) ----
  {
    const _Float16* qkvw = w16 + WS_QKV;          // [384][128]
    for (int t = 0; t < 3; ++t) {
      int nt = wv * 3 + t;                        // 0..23
      v16h bf[4];
#pragma unroll
      for (int ks = 0; ks < 4; ++ks)
        bf[ks] = frag_b(qkvw + (nt * 16) * DIMC + ks * 32, DIMC, lane);
      // tile-uniform decode: which/head from nt, d per lane. scale pre-folded.
      int which = nt >> 3;                        // 0=q 1=k 2=v
      int head  = (nt >> 1) & 3;
      int d     = ((nt & 1) << 4) + nlane;
      int isv   = -(which >> 1);                  // -1 iff v, else 0
      float bias = qkvb_s[nt * 16 + nlane];
      for (int mt = 0; mt < 4; ++mt) {
        v8f acc = {};
#pragma unroll
        for (int ks = 0; ks < 4; ++ks) {
          v16h af = frag_a(xw + (mt * 16) * XS_ + ks * 32, XS_, lane);
          acc = wmma16(af, bf[ks], acc);
        }
#pragma unroll
        for (int r = 0; r < 8; ++r) {
          int m = mt * 16 + r + mofs;
          float v = acc[r] + bias;
          // branchless destination blend: q/k vs transposed v
          int aqk = OFF_QB + (head * 64 + m) * QS_ + d + which * (4 * 64 * QS_);
          int av  = OFF_VT + (head * 32 + d) * VS_ + m;
          int off = aqk ^ ((aqk ^ av) & isv);
          smf[off] = (_Float16)v;
        }
      }
    }
  }
  __syncthreads();

  // ---- stage 3: per-head QK^T + bias + shift mask + softmax ----
  const int h = wv >> 1, sub = wv & 1;
  const bool edge = (wh == 7) || (ww == 7);
  for (int mi = 0; mi < 2; ++mi) {
    int mt = sub * 2 + mi;                        // rows [mt*16, mt*16+16)
    v16h af = frag_a(qb + (h * 64 + mt * 16) * QS_, QS_, lane);
    v8f acc[4];
#pragma unroll
    for (int nt = 0; nt < 4; ++nt) {
      v16h bf = frag_b(kb + (h * 64 + nt * 16) * QS_, QS_, lane);
      v8f z = {};
      acc[nt] = wmma16(af, bf, z);
    }
#pragma unroll
    for (int nt = 0; nt < 4; ++nt) {
      int n = nt * 16 + nlane;
      int kreg = reg3(wh * 8 + (n >> 3)) * 3 + reg3(ww * 8 + (n & 7));
#pragma unroll
      for (int r = 0; r < 8; ++r) {
        int m = mt * 16 + r + mofs;
        float v = acc[nt][r] + biasf[h * 4096 + m * 64 + n];
        if (edge) {
          int qreg = reg3(wh * 8 + (m >> 3)) * 3 + reg3(ww * 8 + (m & 7));
          if (qreg != kreg) v -= 100.0f;
        }
        acc[nt][r] = v;
      }
    }
#pragma unroll
    for (int r = 0; r < 8; ++r) {
      float mx = fmaxf(fmaxf(acc[0][r], acc[1][r]), fmaxf(acc[2][r], acc[3][r]));
#pragma unroll
      for (int s = 1; s < 16; s <<= 1) mx = fmaxf(mx, __shfl_xor(mx, s, 32));
      float e0 = __expf(acc[0][r] - mx), e1 = __expf(acc[1][r] - mx);
      float e2 = __expf(acc[2][r] - mx), e3 = __expf(acc[3][r] - mx);
      float sm = e0 + e1 + e2 + e3;
#pragma unroll
      for (int s = 1; s < 16; s <<= 1) sm += __shfl_xor(sm, s, 32);
      float inv = 1.0f / sm;
      int m = mt * 16 + r + mofs;
      _Float16* arow = at + (h * 64 + m) * AT_;
      arow[0 * 16 + nlane] = (_Float16)(e0 * inv);
      arow[1 * 16 + nlane] = (_Float16)(e1 * inv);
      arow[2 * 16 + nlane] = (_Float16)(e2 * inv);
      arow[3 * 16 + nlane] = (_Float16)(e3 * inv);
    }
  }
  __syncthreads();

  // ---- stage 4: attn @ V per head ----
  for (int mi = 0; mi < 2; ++mi) {
    int mt = sub * 2 + mi;
    v16h af[2];
#pragma unroll
    for (int ks = 0; ks < 2; ++ks)
      af[ks] = frag_a(at + (h * 64 + mt * 16) * AT_ + ks * 32, AT_, lane);
    for (int nd = 0; nd < 2; ++nd) {
      v8f acc = {};
#pragma unroll
      for (int ks = 0; ks < 2; ++ks) {
        v16h bf = frag_b(vT + (h * 32 + nd * 16) * VS_ + ks * 32, VS_, lane);
        acc = wmma16(af[ks], bf, acc);
      }
#pragma unroll
      for (int r = 0; r < 8; ++r) {
        int m = mt * 16 + r + mofs;
        ao[m * AOS_ + h * 32 + nd * 16 + nlane] = (_Float16)acc[r];
      }
    }
  }
  __syncthreads();

  // ---- stage 5: projection 64x128 @ 128x128 ----
  {
    const _Float16* pw = w16 + WS_PROJ;           // [128][128]
    int mt = wv >> 1, ntb = (wv & 1) * 4;
    v16h af[4];
#pragma unroll
    for (int ks = 0; ks < 4; ++ks)
      af[ks] = frag_a(ao + (mt * 16) * AOS_ + ks * 32, AOS_, lane);
    for (int t = 0; t < 4; ++t) {
      int nt = ntb + t;
      v8f acc = {};
#pragma unroll
      for (int ks = 0; ks < 4; ++ks) {
        v16h bf = frag_b(pw + (nt * 16) * DIMC + ks * 32, DIMC, lane);
        acc = wmma16(af[ks], bf, acc);
      }
      int col = nt * 16 + nlane;
      float pb = proj_b[col];
#pragma unroll
      for (int r = 0; r < 8; ++r) {
        int m = mt * 16 + r + mofs;
        po[m * POS_ + col] = acc[r] + pb;
      }
    }
  }
  __syncthreads();

  // ---- stage 6: LayerNorm + residual, scatter to un-rolled coords ----
  {
    int tok = tid >> 2, part = tid & 3;           // 4 threads per token
    const float* row = po + tok * POS_;
    float s = 0.f, ss = 0.f;
#pragma unroll
    for (int i = 0; i < 32; ++i) {
      float v = row[part * 32 + i];
      s += v; ss += v * v;
    }
    s  += __shfl_xor(s, 1, 32);  s  += __shfl_xor(s, 2, 32);
    ss += __shfl_xor(ss, 1, 32); ss += __shfl_xor(ss, 2, 32);
    float mu = s * (1.0f / 128.0f);
    float var = ss * (1.0f / 128.0f) - mu * mu;
    float rs = rsqrtf(var + 1e-5f);
    int gh = wh * 8 + (tok >> 3), gw = ww * 8 + (tok & 7);
    int oh = (gh + 4) & 63, ow = (gw + 4) & 63;
    long base = ((long)(b * 4096 + oh * 64 + ow)) * DIMC;
#pragma unroll
    for (int i = 0; i < 32; ++i) {
      int ch = part * 32 + i;
      float v = (row[ch] - mu) * rs * n1w[ch] + n1b[ch];
      out[base + ch] = x[base + ch] + v;
    }
  }
}

// ---------------- kernel 4: fused MLP + LN + residual (in-place on d_out) --
__global__ __launch_bounds__(256)
void k_mlp_block(const _Float16* __restrict__ w16,
                 const float* __restrict__ fc1_b, const float* __restrict__ fc2_b,
                 const float* __restrict__ n2w, const float* __restrict__ n2b,
                 float* __restrict__ xio) {
  extern __shared__ char smem[];
  float*    x1 = (float*)smem;                    // [64][128] residual copy
  _Float16* xw = (_Float16*)(x1 + 64 * DIMC);     // [64][130]
  _Float16* hb = xw + 64 * XS_;                   // [64][514]
  float*    ob = (float*)(hb + 64 * HS_);         // [64][132]

  const int tid = threadIdx.x, lane = tid & 31, wv = tid >> 5;
  const int nlane = lane & 15, mofs = (lane >> 4) << 3;
  const long row0 = (long)blockIdx.x * 64;

  for (int i = tid; i < 64 * DIMC; i += 256) {
    int m = i >> 7, ch = i & 127;
    float v = xio[(row0 + m) * DIMC + ch];
    x1[m * DIMC + ch] = v;
    xw[m * XS_ + ch] = (_Float16)v;
  }
  __syncthreads();

  const _Float16* w1 = w16 + WS_FC1;              // [512][128]
  const _Float16* w2 = w16 + WS_FC2;              // [128][512]

  // ---- fc1 (128 -> 512) + exact GELU ----
  for (int t = 0; t < 4; ++t) {
    int nt = wv * 4 + t;                          // 32 ntiles over 8 waves
    v16h bf[4];
#pragma unroll
    for (int ks = 0; ks < 4; ++ks)
      bf[ks] = frag_b(w1 + (nt * 16) * DIMC + ks * 32, DIMC, lane);
    int col = nt * 16 + nlane;
    float bb = fc1_b[col];
    for (int mt = 0; mt < 4; ++mt) {
      v8f acc = {};
#pragma unroll
      for (int ks = 0; ks < 4; ++ks) {
        v16h af = frag_a(xw + (mt * 16) * XS_ + ks * 32, XS_, lane);
        acc = wmma16(af, bf[ks], acc);
      }
#pragma unroll
      for (int r = 0; r < 8; ++r) {
        float v = acc[r] + bb;
        v = 0.5f * v * (1.0f + erff(v * 0.70710678118654752f));
        hb[(mt * 16 + r + mofs) * HS_ + col] = (_Float16)v;
      }
    }
  }
  __syncthreads();

  // ---- fc2 (512 -> 128) ----
  {
    int mt = wv >> 1, ntb = (wv & 1) * 4;
    for (int t = 0; t < 4; ++t) {
      int nt = ntb + t;
      v8f acc = {};
#pragma unroll
      for (int ks = 0; ks < 16; ++ks) {
        v16h af = frag_a(hb + (mt * 16) * HS_ + ks * 32, HS_, lane);
        v16h bf = frag_b(w2 + (nt * 16) * 512 + ks * 32, 512, lane);
        acc = wmma16(af, bf, acc);
      }
      int col = nt * 16 + nlane;
      float bb = fc2_b[col];
#pragma unroll
      for (int r = 0; r < 8; ++r)
        ob[(mt * 16 + r + mofs) * POS_ + col] = acc[r] + bb;
    }
  }
  __syncthreads();

  // ---- LayerNorm + residual, write back in place ----
  {
    int tok = tid >> 2, part = tid & 3;
    const float* row = ob + tok * POS_;
    float s = 0.f, ss = 0.f;
#pragma unroll
    for (int i = 0; i < 32; ++i) {
      float v = row[part * 32 + i];
      s += v; ss += v * v;
    }
    s  += __shfl_xor(s, 1, 32);  s  += __shfl_xor(s, 2, 32);
    ss += __shfl_xor(ss, 1, 32); ss += __shfl_xor(ss, 2, 32);
    float mu = s * (1.0f / 128.0f);
    float var = ss * (1.0f / 128.0f) - mu * mu;
    float rs = rsqrtf(var + 1e-5f);
#pragma unroll
    for (int i = 0; i < 32; ++i) {
      int ch = part * 32 + i;
      float v = (row[ch] - mu) * rs * n2w[ch] + n2b[ch];
      xio[(row0 + tok) * DIMC + ch] = x1[tok * DIMC + ch] + v;
    }
  }
}

// ---------------- host launcher -------------------------------------------
extern "C" void kernel_launch(void* const* d_in, const int* in_sizes, int n_in,
                              void* d_out, int out_size, void* d_ws, size_t ws_size,
                              hipStream_t stream) {
  const float* x      = (const float*)d_in[0];
  const float* qkv_w  = (const float*)d_in[1];
  const float* qkv_b  = (const float*)d_in[2];
  const float* proj_w = (const float*)d_in[3];
  const float* proj_b = (const float*)d_in[4];
  const float* rpe_w1 = (const float*)d_in[5];
  const float* rpe_b1 = (const float*)d_in[6];
  const float* rpe_w2 = (const float*)d_in[7];
  const float* n1w    = (const float*)d_in[8];
  const float* n1b    = (const float*)d_in[9];
  const float* fc1_w  = (const float*)d_in[10];
  const float* fc1_b  = (const float*)d_in[11];
  const float* fc2_w  = (const float*)d_in[12];
  const float* fc2_b  = (const float*)d_in[13];
  const float* n2w    = (const float*)d_in[14];
  const float* n2b    = (const float*)d_in[15];
  float* out = (float*)d_out;

  const int nTok = in_sizes[0] / DIMC;            // B * 4096
  const int nWin = nTok / NTOKW;                  // B * 64 windows

  _Float16* w16 = (_Float16*)d_ws;
  float* biasf   = (float*)((char*)d_ws + WS_BIAS_BYTES);
  float* qkvb_s  = (float*)((char*)d_ws + WS_QKVB_BYTES);

  k_convert_weights<<<192, 256, 0, stream>>>(qkv_w, proj_w, fc1_w, fc2_w,
                                             qkv_b, w16, qkvb_s);
  k_rpe_bias<<<16, 256, 0, stream>>>(rpe_w1, rpe_b1, rpe_w2, biasf);

  const size_t smem_attn = (size_t)OFF_PO * 2 + (size_t)64 * POS_ * 4;  // ~152.6 KB
  k_attn_block<<<nWin, 256, smem_attn, stream>>>(x, w16, biasf, qkvb_s, proj_b,
                                                 n1w, n1b, out);

  const size_t smem_mlp = (size_t)64 * DIMC * 4 + (64 * XS_ + 64 * HS_) * 2 +
                          (size_t)64 * POS_ * 4;                        // ~149 KB
  k_mlp_block<<<nTok / 64, 256, smem_mlp, stream>>>(w16, fc1_b, fc2_b,
                                                    n2w, n2b, out);
}
